// LlamaAttention_2027224563896
// MI455X (gfx1250) — compile-verified, matrix operand order checked by reading, not run on hardware
//
#include <hip/hip_runtime.h>

// ---------------------------------------------------------------------------
// MI455X (gfx1250) Llama attention layer.
// ~77 GFLOP vs ~72MB traffic -> compute bound; every matmul goes through
// v_wmma_f32_16x16x32_f16 (f32 accumulate). Global->LDS staging uses the
// CDNA5 async DMA path (global_load_async_to_lds_b128 + ASYNCcnt) with
// double-buffered LDS tiles so DMA overlaps the WMMA pipeline.
// ---------------------------------------------------------------------------

typedef _Float16 h8  __attribute__((ext_vector_type(8)));
typedef _Float16 h16 __attribute__((ext_vector_type(16)));
typedef float    f8  __attribute__((ext_vector_type(8)));

union H16 { h16 v; h8 h[2]; };

// Low 32 bits of a generic pointer into LDS == byte offset in the wave's LDS
// allocation (aperture layout: addr[31:0] is the LDS address).
__device__ __forceinline__ unsigned lds_off(const void* p) {
  return (unsigned)(unsigned long long)p;
}

// CDNA5 async DMA: copy 16B/lane from global to LDS, tracked by ASYNCcnt.
__device__ __forceinline__ void async_ld_b128(unsigned dst_lds,
                                              const void* src_global) {
  asm volatile("global_load_async_to_lds_b128 %0, %1, off"
               :: "v"(dst_lds), "v"(src_global)
               : "memory");
}

__device__ __forceinline__ void wait_async0() {
  asm volatile("s_wait_asynccnt 0" ::: "memory");
}

// CDNA5 16-bit A/B fragment layout (CDNA5 ISA §7.12.2):
//   lane l (0..15) holds row l; K elems [hi*8 .. hi*8+7] and [16+hi*8 .. 23+hi*8]
// Both chunks 16B-contiguous -> two ds_load_b128 per fragment.
__device__ __forceinline__ h16 load_frag_lds(const _Float16* base, int stride,
                                             int l, int hi) {
  H16 r;
  const _Float16* p = base + l * stride + hi * 8;
  r.h[0] = *(const h8*)(p);
  r.h[1] = *(const h8*)(p + 16);
  return r.v;
}

// ---------------------------------------------------------------------------
// f32 -> f16 conversion (2x b128 load, 1x b128 store per thread)
// ---------------------------------------------------------------------------
__global__ __launch_bounds__(256) void f32_to_f16_kernel(
    const float* __restrict__ in, _Float16* __restrict__ out, int n) {
  int i = (blockIdx.x * 256 + threadIdx.x) * 8;
  if (i >= n) return;
  float4 a = *(const float4*)(in + i);
  float4 b = *(const float4*)(in + i + 4);
  h8 o;
  o[0] = (_Float16)a.x; o[1] = (_Float16)a.y;
  o[2] = (_Float16)a.z; o[3] = (_Float16)a.w;
  o[4] = (_Float16)b.x; o[5] = (_Float16)b.y;
  o[6] = (_Float16)b.z; o[7] = (_Float16)b.w;
  *(h8*)(out + i) = o;
}

// ---------------------------------------------------------------------------
// In-place RoPE on f16 Q or K (f32 math). One thread per (s, head, d<32) pair.
// out[d] = x[d]*cos - x[d+32]*sin ; out[d+32] = x[d+32]*cos + x[d]*sin
// ---------------------------------------------------------------------------
__global__ __launch_bounds__(256) void rope_inplace_kernel(
    _Float16* __restrict__ x, const float* __restrict__ cosb,
    const float* __restrict__ sinb, int nheads, int total) {
  int idx = blockIdx.x * 256 + threadIdx.x;
  if (idx >= total) return;
  int d = idx & 31;
  int t = idx >> 5;
  int h = t % nheads;
  int s = t / nheads;
  _Float16* row = x + (size_t)s * (nheads * 64) + h * 64;
  float q1 = (float)row[d];
  float q2 = (float)row[d + 32];
  float c1 = cosb[s * 64 + d],      s1 = sinb[s * 64 + d];
  float c2 = cosb[s * 64 + d + 32], s2 = sinb[s * 64 + d + 32];
  row[d]      = (_Float16)(q1 * c1 - q2 * s1);
  row[d + 32] = (_Float16)(q2 * c2 + q1 * s2);
}

// ---------------------------------------------------------------------------
// Generic GEMM: C[M,N] = A[M,K] * B[N,K]^T (f16 in, f32 accumulate).
// Block tile 128x128, BK=32, 8 waves (2x4), 4x2 WMMA tiles per wave/K-step.
// Async double-buffered LDS staging: stage k+1 DMA overlaps stage k WMMAs.
// EPI: 0 = f32 out, 1 = f16 out, 2 = f16 transposed out (builds V^T).
// ---------------------------------------------------------------------------
template <int EPI>
__global__ __launch_bounds__(256) void gemm_f16_nt_kernel(
    const _Float16* __restrict__ A, const _Float16* __restrict__ B,
    void* __restrict__ Cout, int M, int N, int K, int ldc) {
  constexpr int BM = 128, BN = 128, BK = 32, LDT = BK + 8;  // pad: 40 halves
  __shared__ alignas(16) _Float16 As[2][BM * LDT];
  __shared__ alignas(16) _Float16 Bs[2][BN * LDT];

  const int tid  = threadIdx.x;
  const int lane = tid & 31, l = lane & 15, hi = lane >> 4;
  const int wave = tid >> 5, wm = wave >> 2, wn = wave & 3;
  const int mBase = blockIdx.y * BM, nBase = blockIdx.x * BN;

  // 128x32 halves per operand = 512 b128 chunks; 2 per thread per operand.
  auto issue_stage = [&](int k0, int buf) {
#pragma unroll
    for (int i = 0; i < 2; ++i) {
      int c   = tid + i * 256;
      int row = c >> 2, col = (c & 3) * 8;
      async_ld_b128(lds_off(&As[buf][row * LDT + col]),
                    &A[(size_t)(mBase + row) * K + k0 + col]);
      async_ld_b128(lds_off(&Bs[buf][row * LDT + col]),
                    &B[(size_t)(nBase + row) * K + k0 + col]);
    }
  };

  f8 acc[4][2] = {};
  const int nk = K / BK;
  issue_stage(0, 0);

  for (int ks = 0; ks < nk; ++ks) {
    const int cur = ks & 1;
    wait_async0();        // this wave's stage-ks DMA complete
    __syncthreads();      // all waves' DMA complete -> tile readable
    if (ks + 1 < nk) issue_stage((ks + 1) * BK, cur ^ 1);  // overlap next DMA

    H16 a[4], b[2];
#pragma unroll
    for (int mt = 0; mt < 4; ++mt)
      a[mt].v = load_frag_lds(&As[cur][(wm * 64 + mt * 16) * LDT], LDT, l, hi);
#pragma unroll
    for (int nt = 0; nt < 2; ++nt)
      b[nt].v = load_frag_lds(&Bs[cur][(wn * 32 + nt * 16) * LDT], LDT, l, hi);

#pragma unroll
    for (int mt = 0; mt < 4; ++mt)
#pragma unroll
      for (int nt = 0; nt < 2; ++nt)
        acc[mt][nt] = __builtin_amdgcn_wmma_f32_16x16x32_f16(
            false, a[mt].v, false, b[nt].v, (short)0, acc[mt][nt], false, false);
    __syncthreads();      // done reading buf[cur] before next iter overwrites
  }

  // Epilogue. f32 C/D layout: n = n0 + l ; m = m0 + hi*8 + r (r = vgpr idx).
#pragma unroll
  for (int mt = 0; mt < 4; ++mt)
#pragma unroll
    for (int nt = 0; nt < 2; ++nt) {
      int n    = nBase + wn * 32 + nt * 16 + l;
      int mrow = mBase + wm * 64 + mt * 16 + hi * 8;
#pragma unroll
      for (int r = 0; r < 8; ++r) {
        float v = acc[mt][nt][r];
        int   m = mrow + r;
        if (EPI == 0)
          ((float*)Cout)[(size_t)m * ldc + n] = v;
        else if (EPI == 1)
          ((_Float16*)Cout)[(size_t)m * ldc + n] = (_Float16)v;
        else
          ((_Float16*)Cout)[(size_t)n * ldc + m] = (_Float16)v;
      }
    }
}

// ---------------------------------------------------------------------------
// Flash attention: grid = (32 heads, 32 query-blocks), block = 128 (4 waves).
// Each wave owns 16 queries; keys in blocks of 32 with online softmax.
// K/V^T tiles staged via async DMA with double buffering.
// ---------------------------------------------------------------------------
__global__ __launch_bounds__(128) void flash_attn_kernel(
    const _Float16* __restrict__ Qh, const _Float16* __restrict__ Kh,
    const _Float16* __restrict__ Vt, _Float16* __restrict__ Oh) {
  constexpr int S = 2048, HQ = 2048, HKV = 512;
  constexpr int LQ = 72, LK = 72, LV = 40, LP = 40;
  __shared__ alignas(16) _Float16 Qs[64 * LQ];
  __shared__ alignas(16) _Float16 Ks[2][32 * LK];
  __shared__ alignas(16) _Float16 Vts[2][64 * LV];
  __shared__ alignas(16) _Float16 Ps[4 * 16 * LP];

  const int h = blockIdx.x, qb = blockIdx.y;
  const int kv = h >> 2;  // 4-way GQA repeat
  const int tid = threadIdx.x;
  const int lane = tid & 31, l = lane & 15, hi = lane >> 4;
  const int wave = tid >> 5;

  // Stage the 64x64 Q tile (async, waited together with KV stage 0).
#pragma unroll
  for (int i = 0; i < 4; ++i) {
    int c = tid + i * 128;
    int row = c >> 3, col = (c & 7) * 8;
    async_ld_b128(lds_off(&Qs[row * LQ + col]),
                  &Qh[(size_t)(qb * 64 + row) * HQ + h * 64 + col]);
  }

  // K tile (32 keys x 64 d) + V^T tile (64 d x 32 keys): 4 b128 per thread.
  auto issue_kv = [&](int kb, int buf) {
#pragma unroll
    for (int i = 0; i < 2; ++i) {
      int c = tid + i * 128;
      int row = c >> 3, col = (c & 7) * 8;
      async_ld_b128(lds_off(&Ks[buf][row * LK + col]),
                    &Kh[(size_t)(kb * 32 + row) * HKV + kv * 64 + col]);
      int rv = c >> 2, cv = (c & 3) * 8;
      async_ld_b128(lds_off(&Vts[buf][rv * LV + cv]),
                    &Vt[(size_t)(kv * 64 + rv) * S + kb * 32 + cv]);
    }
  };
  issue_kv(0, 0);

  wait_async0();
  __syncthreads();

  H16 aq[2];
  aq[0].v = load_frag_lds(&Qs[(wave * 16) * LQ + 0], LQ, l, hi);
  aq[1].v = load_frag_lds(&Qs[(wave * 16) * LQ + 32], LQ, l, hi);

  f8 o[4] = {};
  float mrun[8], lrun[8];
#pragma unroll
  for (int r = 0; r < 8; ++r) { mrun[r] = -1e30f; lrun[r] = 0.0f; }
  _Float16* myPs = &Ps[wave * 16 * LP];
  const float sc = 0.125f;  // 1/sqrt(64)
  constexpr int NKB = S / 32;

  for (int kb = 0; kb < NKB; ++kb) {
    const int cur = kb & 1;
    if (kb) {
      wait_async0();
      __syncthreads();
    }
    if (kb + 1 < NKB) issue_kv(kb + 1, cur ^ 1);  // overlap next DMA

    // scores tile: 16 queries x 32 keys (two 16-wide n-tiles)
    f8 s[2] = {};
#pragma unroll
    for (int nt = 0; nt < 2; ++nt) {
      H16 b0, b1;
      b0.v = load_frag_lds(&Ks[cur][(nt * 16) * LK + 0], LK, l, hi);
      b1.v = load_frag_lds(&Ks[cur][(nt * 16) * LK + 32], LK, l, hi);
      s[nt] = __builtin_amdgcn_wmma_f32_16x16x32_f16(
          false, aq[0].v, false, b0.v, (short)0, s[nt], false, false);
      s[nt] = __builtin_amdgcn_wmma_f32_16x16x32_f16(
          false, aq[1].v, false, b1.v, (short)0, s[nt], false, false);
    }

    // Online softmax. Row m = hi*8 + r lives across the 16 lanes of a half.
    float mnew[8], corr[8];
#pragma unroll
    for (int r = 0; r < 8; ++r) {
      float v = fmaxf(s[0][r], s[1][r]) * sc;
#pragma unroll
      for (int mk = 1; mk < 16; mk <<= 1) v = fmaxf(v, __shfl_xor(v, mk, 32));
      mnew[r] = fmaxf(mrun[r], v);
      corr[r] = __expf(mrun[r] - mnew[r]);
      mrun[r] = mnew[r];
    }

    // P = exp(s*sc - mnew); stage P (f16) as next WMMA's A operand.
#pragma unroll
    for (int nt = 0; nt < 2; ++nt)
#pragma unroll
      for (int r = 0; r < 8; ++r) {
        float p = __expf(s[nt][r] * sc - mnew[r]);
        s[nt][r] = p;
        myPs[(hi * 8 + r) * LP + nt * 16 + l] = (_Float16)p;
      }
#pragma unroll
    for (int r = 0; r < 8; ++r) {
      float rs = s[0][r] + s[1][r];
#pragma unroll
      for (int mk = 1; mk < 16; mk <<= 1) rs += __shfl_xor(rs, mk, 32);
      lrun[r] = lrun[r] * corr[r] + rs;
#pragma unroll
      for (int dt = 0; dt < 4; ++dt) o[dt][r] *= corr[r];
    }

    // O += P(16x32) x V(32x64): B operand = V^T rows (n = out dim, k = keys).
    H16 pf;
    pf.v = load_frag_lds(myPs, LP, l, hi);
#pragma unroll
    for (int dt = 0; dt < 4; ++dt) {
      H16 bv;
      bv.v = load_frag_lds(&Vts[cur][(dt * 16) * LV], LV, l, hi);
      o[dt] = __builtin_amdgcn_wmma_f32_16x16x32_f16(
          false, pf.v, false, bv.v, (short)0, o[dt], false, false);
    }
    __syncthreads();  // done reading buf[cur] before next iter overwrites it
  }

  // Normalize and write f16 attention output (row-major [S, 2048]).
#pragma unroll
  for (int dt = 0; dt < 4; ++dt)
#pragma unroll
    for (int r = 0; r < 8; ++r) {
      float v = o[dt][r] / lrun[r];
      int srow = qb * 64 + wave * 16 + hi * 8 + r;
      int col  = h * 64 + dt * 16 + l;
      Oh[(size_t)srow * HQ + col] = (_Float16)v;
    }
}

// ---------------------------------------------------------------------------
// Host-side orchestration (graph-capture safe: stream launches only).
// ---------------------------------------------------------------------------
extern "C" void kernel_launch(void* const* d_in, const int* in_sizes, int n_in,
                              void* d_out, int out_size, void* d_ws,
                              size_t ws_size, hipStream_t stream) {
  (void)in_sizes; (void)n_in; (void)out_size; (void)ws_size;
  const float* hidden = (const float*)d_in[0];
  const float* cosb   = (const float*)d_in[1];
  const float* sinb   = (const float*)d_in[2];
  const float* Wq     = (const float*)d_in[3];
  const float* Wk     = (const float*)d_in[4];
  const float* Wv     = (const float*)d_in[5];
  const float* Wo     = (const float*)d_in[6];
  float* out = (float*)d_out;

  const int S = 2048, H = 2048, KV = 512;

  char* ws = (char*)d_ws;
  _Float16* Xh  = (_Float16*)ws; ws += (size_t)S * H * 2;    // 8 MB
  _Float16* Wqh = (_Float16*)ws; ws += (size_t)H * H * 2;    // 8 MB
  _Float16* Wkh = (_Float16*)ws; ws += (size_t)KV * H * 2;   // 2 MB
  _Float16* Wvh = (_Float16*)ws; ws += (size_t)KV * H * 2;   // 2 MB
  _Float16* Woh = (_Float16*)ws; ws += (size_t)H * H * 2;    // 8 MB
  _Float16* Qh  = (_Float16*)ws; ws += (size_t)S * H * 2;    // 8 MB
  _Float16* Kh  = (_Float16*)ws; ws += (size_t)S * KV * 2;   // 2 MB
  _Float16* Vt  = (_Float16*)ws; ws += (size_t)KV * S * 2;   // 2 MB (V^T)
  _Float16* AOh = (_Float16*)ws; ws += (size_t)S * H * 2;    // 8 MB

  auto cvt = [&](const float* src, _Float16* dst, int n) {
    f32_to_f16_kernel<<<(n / 8 + 255) / 256, 256, 0, stream>>>(src, dst, n);
  };
  cvt(hidden, Xh, S * H);
  cvt(Wq, Wqh, H * H);
  cvt(Wk, Wkh, KV * H);
  cvt(Wv, Wvh, KV * H);
  cvt(Wo, Woh, H * H);

  // Projections: Q [2048x2048], K [2048x512], V^T [512x2048]
  gemm_f16_nt_kernel<1><<<dim3(H / 128, S / 128), 256, 0, stream>>>(
      Xh, Wqh, (void*)Qh, S, H, H, H);
  gemm_f16_nt_kernel<1><<<dim3(KV / 128, S / 128), 256, 0, stream>>>(
      Xh, Wkh, (void*)Kh, S, KV, H, KV);
  gemm_f16_nt_kernel<2><<<dim3(KV / 128, S / 128), 256, 0, stream>>>(
      Xh, Wvh, (void*)Vt, S, KV, H, S);

  // RoPE in place on Q and K.
  {
    int totQ = S * 32 * 32;
    rope_inplace_kernel<<<(totQ + 255) / 256, 256, 0, stream>>>(Qh, cosb, sinb,
                                                                32, totQ);
    int totK = S * 8 * 32;
    rope_inplace_kernel<<<(totK + 255) / 256, 256, 0, stream>>>(Kh, cosb, sinb,
                                                                8, totK);
  }

  // Flash attention: 32 heads x 32 query blocks of 64.
  flash_attn_kernel<<<dim3(32, S / 64), 128, 0, stream>>>(Qh, Kh, Vt, AOh);

  // Output projection -> f32 d_out.
  gemm_f16_nt_kernel<0><<<dim3(H / 128, S / 128), 256, 0, stream>>>(
      AOh, Woh, (void*)out, S, H, H, H);
}
